// ShiftConv2D1_27865747816605
// MI455X (gfx1250) — matrix operand order repaired and checked by reading, not compiled, not run on hardware
//
#include <hip/hip_runtime.h>

// ---------------------------------------------------------------------------
// ShiftConv (ShiftNet 5-way shift + 1x1 conv) for MI455X / gfx1250.
// GEMM: M = 8*256*256 = 524288, K = N = 180 (padded to 192).
// bf16 WMMA (f32 accumulate) keeps the kernel HBM-bound (~32us floor).
// v3: pure global addressing (saddr + int32 voffset, no flat ops),
//     bias via WMMA C-operand (zero-move acc init), branchless staging,
//     batched A-fragment ds_loads.
// ---------------------------------------------------------------------------

typedef __bf16 bf16_t;
typedef __attribute__((ext_vector_type(4)))  bf16_t bf16x4;
typedef __attribute__((ext_vector_type(8)))  bf16_t bf16x8;
typedef __attribute__((ext_vector_type(16))) bf16_t bf16x16;
typedef __attribute__((ext_vector_type(8)))  float  f32x8;

#define BB    8
#define HH    256
#define WW    256
#define CC    180
#define NT    12          // N tiles of 16 (192)
#define KT    6           // K tiles of 32 (192)
#define LDA   200         // LDS row stride in bf16 elems (400B: bank-friendly)
#define TPB   192         // 6 waves
#define REPS  4           // 768 float4-slots / 192 threads
#define NPW   2           // N-tiles per wave

// ---------------------------------------------------------------------------
// Repack W1 (180x180 f32, [c][d]) into fragment-ordered bf16 B-matrix tiles.
// Fragment f = nt*KT + kt; lane L owns 16 contiguous bf16:
//   N = nt*16 + (L&15),  K = kt*32 + (L>=16 ? 16 : 0) + e
// a contiguous K order shared with the A fragments read from LDS (2 K-values per VGPR).
// ---------------------------------------------------------------------------
__global__ __launch_bounds__(256) void repack_w_kernel(
    const float* __restrict__ W1, bf16_t* __restrict__ wfrag) {
  int t = blockIdx.x * 256 + threadIdx.x;
  if (t >= NT * KT * 32) return;
  int lane = t & 31;
  int f    = t >> 5;
  int kt   = f % KT;
  int nt   = f / KT;
  int n    = nt * 16 + (lane & 15);
  int kb   = kt * 32 + ((lane & 16) ? 16 : 0);
  bf16_t* dst = wfrag + (size_t)t * 16;
  #pragma unroll
  for (int e = 0; e < 16; ++e) {
    int k = kb + e;
    float v = (k < CC && n < CC) ? W1[k * CC + n] : 0.0f;
    dst[e] = (bf16_t)v;
  }
}

// ---------------------------------------------------------------------------
// Main kernel: one block per image row (b,h); 6 waves, 2 N-tiles per wave.
// ---------------------------------------------------------------------------
__global__ __launch_bounds__(TPB) void shiftconv_kernel(
    const float* __restrict__ x,      // [B,H,W,C] f32
    const float* __restrict__ W1,     // [C,C] f32 (fallback path)
    const float* __restrict__ b1,     // [C] f32
    const bf16_t* __restrict__ wfrag, // repacked B fragments in d_ws
    float* __restrict__ out,          // [B,H,W,C] f32
    int use_ws) {
  __shared__ bf16_t smA[2][16 * LDA];

  const int tid  = threadIdx.x;
  const int lane = tid & 31;
  const int wv   = tid >> 5;          // wave id (0..5)
  const int bid  = blockIdx.x;        // b*H + h
  const int h    = bid & (HH - 1);
  const int bimg = bid >> 8;
  const int hi   = (lane >> 4) & 1;
  const int ln   = lane & 15;

  // ---- B fragments: 2 N-tiles x 6 K-tiles, kept in VGPRs ----
  bf16x16 bfrag[NPW][KT];
  #pragma unroll
  for (int t = 0; t < NPW; ++t) {
    const int nt = wv * NPW + t;
    if (use_ws) {
      #pragma unroll
      for (int kt = 0; kt < KT; ++kt) {
        const bf16x8* q =
            (const bf16x8*)(wfrag + ((size_t)((nt * KT + kt) * 32 + lane)) * 16);
        bf16x8 lo = q[0];
        bf16x8 hv = q[1];
        bfrag[t][kt] = __builtin_shufflevector(lo, hv, 0, 1, 2, 3, 4, 5, 6, 7,
                                               8, 9, 10, 11, 12, 13, 14, 15);
      }
    } else {
      int n = nt * 16 + ln;
      #pragma unroll
      for (int kt = 0; kt < KT; ++kt) {
        int kb = kt * 32 + hi * 16;
        #pragma unroll
        for (int e = 0; e < 16; ++e) {
          int k = kb + e;
          float v = (k < CC && n < CC) ? W1[k * CC + n] : 0.0f;
          bfrag[t][kt][e] = (bf16_t)v;
        }
      }
    }
  }

  // ---- bias as WMMA C-operand (zero-move accumulator init) ----
  int   ncol[NPW];
  f32x8 biasv[NPW];
  #pragma unroll
  for (int t = 0; t < NPW; ++t) {
    ncol[t] = (wv * NPW + t) * 16 + ln;
    float bv = (ncol[t] < CC) ? b1[ncol[t]] : 0.0f;
    #pragma unroll
    for (int e = 0; e < 8; ++e) biasv[t][e] = bv;
  }
  const bool stok1 = (ncol[NPW - 1] < CC);  // only last tile can be partial

  // ---- loop-invariant A-tile slot decomposition (4 slots per thread) ----
  // slot q = r*192+tid: pixel p = q/48, float4 index c4 = q%48 (45..47 = pad)
  int     goff[REPS];    // running global offset in floats (+16*CC per iter)
  int     w2[REPS];      // running source w coordinate
  bool    alive[REPS];   // h-valid && not a pad slot
  bf16_t* sA0[REPS];     // LDS dest, buffer 0 (buffer 1 = +16*LDA)
  #pragma unroll
  for (int r = 0; r < REPS; ++r) {
    int q  = r * TPB + tid;
    int p  = q / 48;
    int c4 = q - p * 48;
    int c  = c4 * 4;
    int g  = c4 / 9;
    int dh = (g == 2) ? 1 : ((g == 3) ? -1 : 0);
    int dw = (g == 0) ? 1 : ((g == 1) ? -1 : 0);
    int h2 = h + dh;
    bool vh = (unsigned)h2 < HH;
    int h2c = vh ? h2 : 0;
    alive[r] = vh && (c4 < 45);
    w2[r]    = p + dw;                       // w coordinate at it = 0
    goff[r]  = ((bimg * HH + h2c) * WW + w2[r]) * CC + c;
    sA0[r]   = &smA[0][p * LDA + c];
  }

  // ---- running output offsets (row m = hi*8 + j, col = ncol) ----
  int ooff[NPW];
  #pragma unroll
  for (int t = 0; t < NPW; ++t)
    ooff[t] = (bid * WW + hi * 8) * CC + ncol[t];

  const bf16_t* rowp0 = &smA[0][ln * LDA];

  // ---- prefetch tile 0 ----
  float4 vreg[REPS];
  bool   vld[REPS];
  #pragma unroll
  for (int r = 0; r < REPS; ++r) {
    vld[r] = alive[r] && (unsigned)w2[r] < WW;
    if (vld[r]) vreg[r] = *(const float4*)(x + goff[r]);
  }

  #pragma unroll 2
  for (int it = 0; it < WW / 16; ++it) {
    const int buf = it & 1;

    // ---- stage prefetched tile into LDS (branchless select + cvt) ----
    #pragma unroll
    for (int r = 0; r < REPS; ++r) {
      float fx = vld[r] ? vreg[r].x : 0.0f;
      float fy = vld[r] ? vreg[r].y : 0.0f;
      float fz = vld[r] ? vreg[r].z : 0.0f;
      float fw = vld[r] ? vreg[r].w : 0.0f;
      bf16x4 cv;
      cv[0] = (bf16_t)fx; cv[1] = (bf16_t)fy;
      cv[2] = (bf16_t)fz; cv[3] = (bf16_t)fw;
      *(bf16x4*)(sA0[r] + buf * (16 * LDA)) = cv;
    }
    __syncthreads();

    // ---- issue global loads for tile it+1 (hidden behind WMMAs below) ----
    if (it + 1 < WW / 16) {
      #pragma unroll
      for (int r = 0; r < REPS; ++r) {
        w2[r]   += 16;
        goff[r] += 16 * CC;
        vld[r] = alive[r] && (unsigned)w2[r] < WW;
        if (vld[r]) vreg[r] = *(const float4*)(x + goff[r]);
      }
    }

    // ---- load all A fragments, then one WMMA burst ----
    const bf16_t* rowp = rowp0 + buf * (16 * LDA);
    bf16x16 af[KT];
    #pragma unroll
    for (int kt = 0; kt < KT; ++kt) {
      int kb = kt * 32 + hi * 16;
      bf16x8 a0 = *(const bf16x8*)(rowp + kb);
      bf16x8 a1 = *(const bf16x8*)(rowp + kb + 8);
      af[kt] = __builtin_shufflevector(a0, a1, 0, 1, 2, 3, 4, 5, 6, 7,
                                       8, 9, 10, 11, 12, 13, 14, 15);
    }

    f32x8 acc[NPW];
    #pragma unroll
    for (int t = 0; t < NPW; ++t) {
      acc[t] = __builtin_amdgcn_wmma_f32_16x16x32_bf16(
          false, af[0], false, bfrag[t][0], (short)0, biasv[t], false, false);
    }
    #pragma unroll
    for (int kt = 1; kt < KT; ++kt) {
      #pragma unroll
      for (int t = 0; t < NPW; ++t) {
        acc[t] = __builtin_amdgcn_wmma_f32_16x16x32_bf16(
            false, af[kt], false, bfrag[t][kt], (short)0, acc[t], false, false);
      }
    }

    // ---- store 2 16x16 tiles (global saddr + int32 offset + imm) ----
    #pragma unroll
    for (int j = 0; j < 8; ++j) out[ooff[0] + j * CC] = acc[0][j];
    if (stok1) {
      #pragma unroll
      for (int j = 0; j < 8; ++j) out[ooff[1] + j * CC] = acc[1][j];
    }
    #pragma unroll
    for (int t = 0; t < NPW; ++t) ooff[t] += 16 * CC;
    // Single barrier/iter is safe with the 2-deep LDS double buffer
    // (every barrier waits DScnt=0, so prior reads are complete before
    //  any wave can overwrite that buffer two iterations later).
  }
}

// ---------------------------------------------------------------------------
extern "C" void kernel_launch(void* const* d_in, const int* in_sizes, int n_in,
                              void* d_out, int out_size, void* d_ws, size_t ws_size,
                              hipStream_t stream) {
  const float* x   = (const float*)d_in[0];  // inputs [8,256,256,180]
  // d_in[1] = mask (unused: shift folded into addressing)
  const float* W1  = (const float*)d_in[2];  // [1,1,180,180]
  const float* b1  = (const float*)d_in[3];  // [180]
  float*       out = (float*)d_out;

  bf16_t* wfrag = (bf16_t*)d_ws;
  const size_t need = (size_t)NT * KT * 32 * 16 * sizeof(bf16_t); // 73728 B
  int use_ws = (ws_size >= need) ? 1 : 0;

  if (use_ws) {
    repack_w_kernel<<<(NT * KT * 32 + 255) / 256, 256, 0, stream>>>(W1, wfrag);
  }
  shiftconv_kernel<<<BB * HH, TPB, 0, stream>>>(x, W1, b1, wfrag, out, use_ws);
}
